// RolloutModule_7687991460005
// MI455X (gfx1250) — matrix operand order repaired and checked by reading, not compiled, not run on hardware
//
#include <hip/hip_runtime.h>

typedef _Float16 h16;
typedef __attribute__((ext_vector_type(16))) _Float16 v16h;
typedef __attribute__((ext_vector_type(8)))  float    v8f;
typedef __attribute__((ext_vector_type(4)))  unsigned int v4u;
typedef __attribute__((ext_vector_type(8)))  int      v8i_t;
typedef __attribute__((ext_vector_type(4)))  int      v4i_t;

#define DEVFN __device__ __forceinline__

namespace {

constexpr int   kBatch   = 32768;
constexpr int   kT       = 64;
constexpr int   kSub     = 4;
constexpr float kDT      = 0.25f;
constexpr float kSpeed   = 2.0f;
constexpr float kWmin    = -4.0f;
constexpr float kWmax    = 10.0f;
constexpr float kGoalTol = 0.45f;
constexpr float kBetaS   = 50.0f;   // spatial smooth-min
constexpr float kBetaT   = 8.0f;    // temporal smooth min/max
constexpr int   kHid     = 256;
constexpr int   kRows    = 128;     // batch rows per workgroup
constexpr int   kThreads = 256;     // 8 wave32s
constexpr int   kApitch  = 264;     // halves per activation row (bank-conflict-free)
constexpr int   kSpitch  = 40;      // halves per staged weight row (= 32 + TDM pad)
constexpr float kWS      = 10.0f;
constexpr float kGoalX   = 4.0f;
constexpr float kGoalY   = 3.0f;

// d_ws layout (bytes); total 286,720 B of scratch used.
constexpr size_t OFF_WT1  = 0;                 // 256 x 32  f16 (w1^T, k-padded)
constexpr size_t OFF_WT2  = 16384;             // 256 x 256 f16 (w2^T)
constexpr size_t OFF_WT3  = 16384 + 131072;    // 256 x 256 f16 (w3^T)
constexpr size_t OFF_WTMU = 147456 + 131072;   // 16  x 256 f16 (wmu^T, n-padded)

DEVFN float clampf(float x, float lo, float hi) { return fminf(fmaxf(x, lo), hi); }

union FragH { v16h v; uint4 q[2]; };

// ---------------------------------------------------------------------------
// TDM staging: DMA a [nrows x 32-half] slab of W^T (row pitch Ksrc halves,
// K-block kb) from global into LDS. The descriptor's pad feature inserts
// 4 DWORDs (8 halves) after every 16 DWORDs (32 halves = one row), which
// reproduces the bank-conflict-free kSpitch=40 LDS layout in hardware.
// Issued by one wave; tracked with TENSORcnt. (ISA 08_async_tensor §7/§8)
// ---------------------------------------------------------------------------
DEVFN void tdm_stage(const h16* __restrict__ src, int Ksrc, int kb, int nrows,
                     h16* s_wt)
{
  const unsigned long long ga =
      (unsigned long long)(uintptr_t)src + (unsigned long long)kb * 64ull;
  const unsigned int lds = (unsigned int)(uintptr_t)s_wt;  // LDS offset (flat low32)

  v4u g0;
  g0[0] = 1u;                                          // count=1, user descriptor
  g0[1] = lds;                                         // lds_addr
  g0[2] = (unsigned int)(ga & 0xffffffffull);          // global_addr[31:0]
  g0[3] = (unsigned int)((ga >> 32) & 0x01ffffffull)   // global_addr[56:32]
          | (2u << 30);                                // type = 2 ("image")

  v8i_t g1;
  g1[0] = (1 << 16)        // data_size = 1 -> 2-byte elements
        | (1 << 20)        // pad_enable (load only)
        | (3 << 22)        // pad_interval: 3 -> every 16 DWORDs (64 B = one row)
        | (3 << 25);       // pad_amount:   3 -> 4 DWORDs (16 B = 8 halves)
  g1[1] = (32 & 0xffff) << 16;               // tensor_dim0[15:0] = 32
  g1[2] = ((nrows & 0xffff) << 16);          // dim0[31:16]=0 | tensor_dim1[15:0]
  g1[3] = (32 << 16);                        // dim1[31:16]=0 | tile_dim0 = 32
  g1[4] = (nrows & 0xffff);                  // tile_dim1 = nrows | tile_dim2 = 0
  g1[5] = Ksrc;                              // tensor_dim0_stride[31:0]
  g1[6] = 0;                                 // stride[47:32] | dim1_stride[15:0]
  g1[7] = 0;

  v4i_t z4 = {0, 0, 0, 0};                   // groups 2/3 unused (<=2-D tensor)
  v8i_t z8 = {0, 0, 0, 0, 0, 0, 0, 0};
  __builtin_amdgcn_tensor_load_to_lds(g0, g1, z4, z4, z8, 0);
}

// One MLP layer on a 128-row activation tile.
//   A (per wave): rows [16*wave, 16*wave+16), all K, from s_act
//   B: TDM-staged weight slab in s_wt, NT 16-column tiles
//   KB: number of 32-wide K blocks (K = 32*KB)
template<int KB, int NT, bool RELU, bool TOMU>
DEVFN void mlp_layer(const h16* __restrict__ wsrc, int Ksrc,
                     const float* __restrict__ bias,
                     h16* s_act, h16* s_wt, float* s_mu, int tid)
{
  const int lane = tid & 31;
  const int wave = tid >> 5;
  const int lm   = lane & 15;
  const int half = lane >> 4;
  const int m    = wave * 16 + lm;

  v8f acc[NT];
  #pragma unroll
  for (int nt = 0; nt < NT; ++nt)
    #pragma unroll
    for (int r = 0; r < 8; ++r) acc[nt][r] = 0.0f;

  for (int kb = 0; kb < KB; ++kb) {
    __syncthreads();                          // stage-buffer reuse fence
    if (tid < 32) {                           // wave 0: EXEC all-ones, issue DMA
      tdm_stage(wsrc, Ksrc, kb, NT * 16, s_wt);
      __builtin_amdgcn_s_wait_tensorcnt(0);   // data resident in LDS
    }
    if (kb + 1 < KB && tid < NT * 16)         // pull next slab toward L2
      __builtin_prefetch(wsrc + (size_t)tid * Ksrc + (kb + 1) * 32, 0, 1);
    __syncthreads();                          // publish slab to all waves

    // A fragment per ISA 16-bit A 16x32 layout:
    // lanes 0-15 k=[0..8)+[16..24), lanes 16-31 k=[8..16)+[24..32)
    FragH A;
    const h16* ap = s_act + m * kApitch + kb * 32 + half * 8;
    A.q[0] = *reinterpret_cast<const uint4*>(ap);
    A.q[1] = *reinterpret_cast<const uint4*>(ap + 16);

    #pragma unroll
    for (int nt = 0; nt < NT; ++nt) {
      // B fragment: lane holds column n = nt*16+lm, k=[half*8..+8)+[16+half*8..+8)
      FragH Bf;
      const h16* bp = s_wt + (nt * 16 + lm) * kSpitch + half * 8;
      Bf.q[0] = *reinterpret_cast<const uint4*>(bp);
      Bf.q[1] = *reinterpret_cast<const uint4*>(bp + 16);
      acc[nt] = __builtin_amdgcn_wmma_f32_16x16x32_f16(
          false, A.v, false, Bf.v, (short)0, acc[nt], false, false);
    }
  }

  if constexpr (TOMU) {
    // D layout: lane -> (n = lm, rows r + 8*half). Only columns 0,1 are real.
    if (lm < 2) {
      #pragma unroll
      for (int r = 0; r < 8; ++r)
        s_mu[(wave * 16 + r + 8 * half) * 2 + lm] = acc[0][r];
    }
  } else {
    #pragma unroll
    for (int nt = 0; nt < NT; ++nt) {
      const int n = nt * 16 + lm;
      const float b = bias[n];
      #pragma unroll
      for (int r = 0; r < 8; ++r) {
        float v = acc[nt][r] + b;
        if (RELU) v = fmaxf(v, 0.0f);
        s_act[(wave * 16 + r + 8 * half) * kApitch + n] = (h16)v;
      }
    }
  }
}

} // namespace

// ---- weight preprocessing: f32 row-major (K,N) -> f16 transposed (N,K) -----
__global__ void prep_weights(const float* __restrict__ w1, const float* __restrict__ w2,
                             const float* __restrict__ w3, const float* __restrict__ wmu,
                             h16* __restrict__ wt1, h16* __restrict__ wt2,
                             h16* __restrict__ wt3, h16* __restrict__ wtmu)
{
  int i = blockIdx.x * blockDim.x + threadIdx.x;
  if (i < 8192) {                              // wt1: [256 n][32 k], k>=13 zero pad
    const int n = i >> 5, k = i & 31;
    wt1[i] = (h16)((k < 13) ? w1[k * 256 + n] : 0.0f);
    return;
  }
  i -= 8192;
  if (i < 65536) {                             // wt2: [256 n][256 k]
    const int n = i >> 8, k = i & 255;
    wt2[i] = (h16)w2[k * 256 + n];
    return;
  }
  i -= 65536;
  if (i < 65536) {                             // wt3
    const int n = i >> 8, k = i & 255;
    wt3[i] = (h16)w3[k * 256 + n];
    return;
  }
  i -= 65536;
  if (i < 4096) {                              // wtmu: [16 n][256 k], n>=2 zero pad
    const int n = i >> 8, k = i & 255;
    wtmu[i] = (h16)((n < 2) ? wmu[k * 2 + n] : 0.0f);
  }
}

// ---- main rollout: 128 agents per workgroup, 64 closed-loop steps ----------
__global__ __launch_bounds__(kThreads)
void rollout_kernel(const float* __restrict__ pos0, const float* __restrict__ wind,
                    const h16* __restrict__ wt1, const h16* __restrict__ wt2,
                    const h16* __restrict__ wt3, const h16* __restrict__ wtmu,
                    const float* __restrict__ b1, const float* __restrict__ b2,
                    const float* __restrict__ b3, const float* __restrict__ bmu,
                    float* __restrict__ out)
{
  __shared__ __align__(16) h16   s_act[kRows * kApitch];  // 66.0 KB activations
  __shared__ __align__(16) h16   s_wt [kHid * kSpitch];   // 20.0 KB weight slab
  __shared__ __align__(16) float s_mu [kRows * 2];        //  1.0 KB policy mean

  const int tid  = threadIdx.x;
  const int row0 = blockIdx.x * kRows;

  const float oxs[3] = {1.75f, 1.75f, 3.75f};
  const float oys[3] = {1.75f, 3.75f, 2.0f};
  const float ors[3] = {0.38f, 0.42f, 0.34f};

  float px = 0.f, py = 0.f, wx = 0.f, wy = 0.f;
  float mSafe = -3.0e38f, sSafe = 0.f;    // online logsumexp of -beta*safe
  float mReach = -3.0e38f, sReach = 0.f;  // online logsumexp of  beta*reach
  if (tid < kRows) {
    const int r = row0 + tid;
    px = pos0[2 * r]; py = pos0[2 * r + 1];
    wx = wind[2 * r]; wy = wind[2 * r + 1];
  }
  const float bmu0 = bmu[0], bmu1 = bmu[1];
  const float sdt  = kDT / (float)kSub;

  #pragma unroll 1
  for (int t = 0; t < kT; ++t) {
    // ---- observation (13 features, zero-padded to K=32) into s_act --------
    if (tid < kRows) {
      h16* o = s_act + tid * kApitch;
      o[0] = (h16)(px / kWS);
      o[1] = (h16)(py / kWS);
      o[2] = (h16)((kGoalX - px) / kWS);
      o[3] = (h16)((kGoalY - py) / kWS);
      #pragma unroll
      for (int i = 0; i < 3; ++i) {
        const float dx = px - oxs[i], dy = py - oys[i];
        const float dist = sqrtf(dx * dx + dy * dy + 1e-9f);
        o[4 + 2 * i] = (h16)(-dx / kWS);
        o[5 + 2 * i] = (h16)(-dy / kWS);
        o[10 + i]    = (h16)(dist - ors[i]);
      }
      #pragma unroll
      for (int i = 13; i < 32; ++i) o[i] = (h16)0.0f;
    }

    // ---- policy MLP (WMMA + TDM weight staging) ---------------------------
    mlp_layer<1, 16, true,  false>(wt1,  32,  b1,      s_act, s_wt, s_mu, tid);
    mlp_layer<8, 16, true,  false>(wt2,  256, b2,      s_act, s_wt, s_mu, tid);
    mlp_layer<8, 16, true,  false>(wt3,  256, b3,      s_act, s_wt, s_mu, tid);
    mlp_layer<8, 1,  false, true >(wtmu, 256, nullptr, s_act, s_wt, s_mu, tid);
    __syncthreads();

    // ---- integrate + STREL margins (per-agent scalar work) -----------------
    if (tid < kRows) {
      const float ax = clampf(s_mu[2 * tid]     + bmu0, -1.f, 1.f);
      const float ay = clampf(s_mu[2 * tid + 1] + bmu1, -1.f, 1.f);
      const float vx = sdt * (kSpeed * ax + wx);
      const float vy = sdt * (kSpeed * ay + wy);
      #pragma unroll
      for (int s = 0; s < kSub; ++s) {
        px = clampf(px + vx, kWmin, kWmax);
        py = clampf(py + vy, kWmin, kWmax);
      }

      // spatial smooth-min clearance, beta=50 (max-shifted)
      float y0, y1, y2;
      {
        float c[3];
        #pragma unroll
        for (int i = 0; i < 3; ++i) {
          const float dx = px - oxs[i], dy = py - oys[i];
          c[i] = sqrtf(dx * dx + dy * dy + 1e-9f) - ors[i];
        }
        y0 = -kBetaS * c[0]; y1 = -kBetaS * c[1]; y2 = -kBetaS * c[2];
      }
      const float Ms = fmaxf(y0, fmaxf(y1, y2));
      const float safeMargin =
          -(Ms + logf(expf(y0 - Ms) + expf(y1 - Ms) + expf(y2 - Ms))) / kBetaS;

      const float gdx = px - kGoalX, gdy = py - kGoalY;
      const float gd = sqrtf(gdx * gdx + gdy * gdy + 1e-9f);
      const float reachMargin = kGoalTol - gd;

      // streaming (online) logsumexp over time, beta=8
      const float xs = -kBetaT * safeMargin;
      if (xs > mSafe) { sSafe = sSafe * expf(mSafe - xs) + 1.0f; mSafe = xs; }
      else            { sSafe += expf(xs - mSafe); }
      const float xr = kBetaT * reachMargin;
      if (xr > mReach) { sReach = sReach * expf(mReach - xr) + 1.0f; mReach = xr; }
      else             { sReach += expf(xr - mReach); }
    }
    __syncthreads();
  }

  // ---- final STREL combine: And(Always(safe), Eventually(reach)) ----------
  if (tid < kRows) {
    const float rhoSafe  = -(mSafe  + logf(sSafe))  / kBetaT;
    const float rhoReach =  (mReach + logf(sReach)) / kBetaT;
    const float a0 = -kBetaT * rhoSafe;
    const float a1 = -kBetaT * rhoReach;
    const float Mp = fmaxf(a0, a1);
    const float rho = -(Mp + logf(expf(a0 - Mp) + expf(a1 - Mp))) / kBetaT;
    out[row0 + tid] = rho;
  }
}

extern "C" void kernel_launch(void* const* d_in, const int* in_sizes, int n_in,
                              void* d_out, int out_size, void* d_ws, size_t ws_size,
                              hipStream_t stream)
{
  const float* pos0 = (const float*)d_in[0];
  const float* wind = (const float*)d_in[1];
  const float* w1   = (const float*)d_in[2];
  const float* b1   = (const float*)d_in[3];
  const float* w2   = (const float*)d_in[4];
  const float* b2   = (const float*)d_in[5];
  const float* w3   = (const float*)d_in[6];
  const float* b3   = (const float*)d_in[7];
  const float* wmu  = (const float*)d_in[8];
  const float* bmu  = (const float*)d_in[9];

  char* ws  = (char*)d_ws;
  h16* wt1  = (h16*)(ws + OFF_WT1);
  h16* wt2  = (h16*)(ws + OFF_WT2);
  h16* wt3  = (h16*)(ws + OFF_WT3);
  h16* wtmu = (h16*)(ws + OFF_WTMU);

  // 143,360 transposed f16 weight elements
  prep_weights<<<560, 256, 0, stream>>>(w1, w2, w3, wmu, wt1, wt2, wt3, wtmu);
  rollout_kernel<<<kBatch / kRows, kThreads, 0, stream>>>(
      pos0, wind, wt1, wt2, wt3, wtmu, b1, b2, b3, bmu, (float*)d_out);
}
